// GPSLayer_58265526337919
// MI455X (gfx1250) — compile-verified, matrix operand order checked by reading, not run on hardware
//
#include <hip/hip_runtime.h>

#define DEV __device__ __forceinline__

typedef __attribute__((ext_vector_type(16))) __bf16 v16bf;
typedef __attribute__((ext_vector_type(8)))  float  v8f;

static constexpr int N_ = 4096;
static constexpr int D_ = 512;
static constexpr int H_ = 8;

DEV unsigned short f2bf(float f) {
  union { float f; unsigned u; } c; c.f = f;
  unsigned u = c.u;
  return (unsigned short)((u + 0x7FFFu + ((u >> 16) & 1u)) >> 16);
}

// Load one 16x32 bf16 WMMA fragment (A-layout; identical mapping serves the
// B operand when "idx" indexes the output column = weight row, contiguous in K).
// lanes 0-15: row=lane,  K = kbase+[0..7] and kbase+16+[0..7]
// lanes 16-31: row=lane-16, K = kbase+8+[0..7] and kbase+24+[0..7]
DEV v16bf load_frag(const unsigned short* p, int stride, int idx, int kbase, int hi) {
  const unsigned short* q = p + (size_t)idx * stride + kbase + hi * 8;
  union { uint4 u[2]; v16bf v; } U;
  U.u[0] = *(const uint4*)(q);
  U.u[1] = *(const uint4*)(q + 16);
  return U.v;
}

DEV v8f wmma_bf(v16bf a, v16bf b, v8f c) {
  return __builtin_amdgcn_wmma_f32_16x16x32_bf16(false, a, false, b, (short)0, c, false, false);
}

DEV float half_max(float v) {
#pragma unroll
  for (int m = 8; m >= 1; m >>= 1) v = fmaxf(v, __shfl_xor(v, m, 32));
  return v;
}
DEV float half_sum(float v) {
#pragma unroll
  for (int m = 8; m >= 1; m >>= 1) v += __shfl_xor(v, m, 32);
  return v;
}

// ---- CDNA5 async global->LDS copy (ASYNCcnt) -----------------------------
DEV unsigned lds_off(const void* p) { return (unsigned)(size_t)p; }

DEV void async_copy_b128(unsigned lds_dst, const void* gsrc) {
  asm volatile("global_load_async_to_lds_b128 %0, %1, off"
               :: "v"(lds_dst), "v"(gsrc) : "memory");
}
DEV void wait_async_le1() { asm volatile("s_wait_asynccnt 0x1" ::: "memory"); }
DEV void wait_async_0()   { asm volatile("s_wait_asynccnt 0x0" ::: "memory"); }

// ---------------------------------------------------------------- utilities
__global__ void zero_kernel(float* p, long long n) {
  long long i = (long long)blockIdx.x * blockDim.x + threadIdx.x;
  if (i < n) p[i] = 0.f;
}

__global__ void cvt_kernel(const float* in, unsigned short* out, int n) {
  int i = blockIdx.x * blockDim.x + threadIdx.x;
  if (i < n) out[i] = f2bf(in[i]);
}

// ------------------------------------------------------------- layer norm
__global__ __launch_bounds__(128) void ln_kernel(
    const float* __restrict__ x,
    const float* g1, const float* b1, float* of1, unsigned short* ob1,
    const float* g2, const float* b2, unsigned short* ob2) {
  __shared__ float red[128];
  int row = blockIdx.x, t = threadIdx.x;
  const float* xr = x + (size_t)row * D_;
  float v[4]; float s = 0.f;
#pragma unroll
  for (int i = 0; i < 4; i++) { v[i] = xr[t * 4 + i]; s += v[i]; }
  red[t] = s; __syncthreads();
  for (int o = 64; o > 0; o >>= 1) { if (t < o) red[t] += red[t + o]; __syncthreads(); }
  float mu = red[0] * (1.f / D_);
  __syncthreads();
  float q = 0.f;
#pragma unroll
  for (int i = 0; i < 4; i++) { float d = v[i] - mu; q += d * d; }
  red[t] = q; __syncthreads();
  for (int o = 64; o > 0; o >>= 1) { if (t < o) red[t] += red[t + o]; __syncthreads(); }
  float inv = rsqrtf(red[0] * (1.f / D_) + 1e-5f);
#pragma unroll
  for (int i = 0; i < 4; i++) {
    int c = t * 4 + i; size_t idx = (size_t)row * D_ + c;
    float nv = (v[i] - mu) * inv;
    float y1 = nv * g1[c] + b1[c];
    if (of1) of1[idx] = y1;
    if (ob1) ob1[idx] = f2bf(y1);
    if (g2) { float y2 = nv * g2[c] + b2[c]; ob2[idx] = f2bf(y2); }
  }
}

// ----------------------------------------------------------------- GEMM
// C[m,o] = sum_k A[m,k]*W[o,k] (+bias[o]) (+residual[m,o]) with epilogue:
// mode 0: none; mode 1: exact GELU; mode 2: scale cols o<D_ by 0.125 (q of qkv).
// Block: 8 waves, tile 32(M) x 256(O); each wave does one 16x64 strip.
// The shared 32x32 A tile is staged through LDS with double-buffered
// global_load_async_to_lds_b128 (ASYNCcnt); B streams straight from global.
__global__ __launch_bounds__(256) void gemm_bf16_kernel(
    const unsigned short* __restrict__ A, int lda,
    const unsigned short* __restrict__ W, int ldw,
    const float* __restrict__ bias, const float* __restrict__ residual,
    float* Cf, unsigned short* Cb, int ldc, int K, int mode) {
  __shared__ __align__(16) unsigned short atile[2][32 * 32];
  int tid = threadIdx.x;
  int lane = tid & 31, wave = tid >> 5;
  int hi = lane >> 4, rc = lane & 15;
  int mrow = (wave >> 2) * 16;                 // local M offset of this wave
  int m0 = blockIdx.y * 32;                    // block M base
  int ob = blockIdx.x * 256 + (wave & 3) * 64; // wave O base

  v8f z = {0.f, 0.f, 0.f, 0.f, 0.f, 0.f, 0.f, 0.f};
  v8f acc[4] = {z, z, z, z};

  int row = tid >> 2, seg = tid & 3;           // copy assignment for tid<128
  const unsigned short* asrc0 = A + (size_t)(m0 + row) * lda + seg * 8;

  int nk = K / 32;
  // prologue: stage tile 0
  if (tid < 128)
    async_copy_b128(lds_off(&atile[0][row * 32 + seg * 8]), asrc0);

  for (int i = 0; i < nk; i++) {
    int kb = i * 32;
    if (i + 1 < nk) {
      if (tid < 128)
        async_copy_b128(lds_off(&atile[(i + 1) & 1][row * 32 + seg * 8]),
                        asrc0 + (kb + 32));
      wait_async_le1();   // tile i landed; tile i+1 may stay in flight
    } else {
      wait_async_0();
    }
    __syncthreads();
    v16bf a = load_frag(atile[i & 1], 32, mrow + rc, 0, hi);
#pragma unroll
    for (int t = 0; t < 4; t++) {
      v16bf b = load_frag(W, ldw, ob + t * 16 + rc, kb, hi);
      acc[t] = wmma_bf(a, b, acc[t]);
    }
    __syncthreads();      // protect buffer before it is overwritten
  }

#pragma unroll
  for (int t = 0; t < 4; t++) {
    int o = ob + t * 16 + rc;
    float bv = bias ? bias[o] : 0.f;
    float scale = (mode == 2 && o < D_) ? 0.125f : 1.f;
#pragma unroll
    for (int r = 0; r < 8; r++) {
      int mm = m0 + mrow + r + 8 * hi;
      size_t idx = (size_t)mm * ldc + o;
      float v = acc[t][r] + bv;
      if (residual) v += residual[idx];
      if (mode == 1) v = 0.5f * v * (1.f + erff(v * 0.70710678118f));
      v *= scale;
      if (Cf) Cf[idx] = v;
      if (Cb) Cb[idx] = f2bf(v);
    }
  }
}

// --------------------------------------------------------- edge scatter
__global__ void deg_kernel(const int* __restrict__ dst, float* deg, int E) {
  int e = blockIdx.x * blockDim.x + threadIdx.x;
  if (e < E) atomicAdd(&deg[dst[e]], 1.f);
}

__global__ void scatter_kernel(const int* __restrict__ src, const int* __restrict__ dst,
                               const float* __restrict__ Al, float* aggr, long long ED) {
  long long i = (long long)blockIdx.x * blockDim.x + threadIdx.x;
  if (i >= ED) return;
  int e = (int)(i >> 9), d = (int)(i & 511);
  atomicAdd(&aggr[(size_t)dst[e] * D_ + d], Al[(size_t)src[e] * D_ + d]);
}

// aggr[n] = (scatterA[n] + deg[n]*(Br[n]+msg_b)) / max(deg,1); Br already has msg_b bias.
__global__ void aggr_fin_kernel(const float* __restrict__ aggr, const float* __restrict__ Br,
                                const float* __restrict__ deg, unsigned short* out, int ND) {
  int i = blockIdx.x * blockDim.x + threadIdx.x;
  if (i >= ND) return;
  int n = i >> 9;
  float dg = deg[n];
  out[i] = f2bf((aggr[i] + dg * Br[i]) / fmaxf(dg, 1.f));
}

// ------------------------------------------------------------------ GRU
__global__ void gru_kernel(const float* __restrict__ gi, const float* __restrict__ gh,
                           const float* __restrict__ xn1, float* xm, int ND) {
  int i = blockIdx.x * blockDim.x + threadIdx.x;
  if (i >= ND) return;
  int n = i >> 9, d = i & 511;
  size_t b = (size_t)n * (3 * D_) + d;
  float r = 1.f / (1.f + __expf(-(gi[b] + gh[b])));
  float zz = 1.f / (1.f + __expf(-(gi[b + D_] + gh[b + D_])));
  float nn = tanhf(gi[b + 2 * D_] + r * gh[b + 2 * D_]);
  xm[i] = (1.f - zz) * nn + zz * xn1[i];
}

// -------------------------------------------------- V transpose ([H*64, N])
__global__ void vt_kernel(const unsigned short* __restrict__ qkv, unsigned short* vT, int total) {
  int i = blockIdx.x * blockDim.x + threadIdx.x;
  if (i >= total) return;
  int n = i & (N_ - 1);
  int hd = i >> 12;  // 0..511
  vT[i] = qkv[(size_t)n * (3 * D_) + 2 * D_ + hd];
}

// -------------------------------------------------------------- attention
// One wave per (16-row tile, head). qkv bf16 [N,1536] with q pre-scaled by 1/8.
// Pass 1: streaming online max/sum over 256 key tiles (WMMA QK^T).
// Pass 2: recompute scores, p -> atomicAdd(p/8) into attn_out, stage P bf16 in
// LDS, WMMA P@V (Vt layout => contiguous B fragments).
__global__ __launch_bounds__(32) void attn_kernel(
    const unsigned short* __restrict__ qkv, const unsigned short* __restrict__ vT,
    float* attn_out, unsigned short* ctx) {
  __shared__ __align__(16) unsigned short pbuf[16 * 32];
  int lane = threadIdx.x;
  int hi = lane >> 4, rc = lane & 15;
  int m0 = blockIdx.x * 16;
  int h = blockIdx.y;
  const unsigned short* qb = qkv + (size_t)m0 * (3 * D_) + h * 64;
  const unsigned short* kb = qkv + D_ + h * 64;
  v16bf aq0 = load_frag(qb, 3 * D_, rc, 0, hi);
  v16bf aq1 = load_frag(qb, 3 * D_, rc, 32, hi);
  const v8f vz = {0.f, 0.f, 0.f, 0.f, 0.f, 0.f, 0.f, 0.f};
  float rmax[8], rsum[8];
#pragma unroll
  for (int r = 0; r < 8; r++) { rmax[r] = -1e30f; rsum[r] = 0.f; }

  for (int j = 0; j < N_ / 16; j++) {
    v8f s = vz;
    s = wmma_bf(aq0, load_frag(kb, 3 * D_, j * 16 + rc, 0, hi), s);
    s = wmma_bf(aq1, load_frag(kb, 3 * D_, j * 16 + rc, 32, hi), s);
#pragma unroll
    for (int r = 0; r < 8; r++) {
      float tmax = half_max(s[r]);
      float nm = fmaxf(rmax[r], tmax);
      float es = half_sum(__expf(s[r] - nm));
      rsum[r] = rsum[r] * __expf(rmax[r] - nm) + es;
      rmax[r] = nm;
    }
  }
  float rinv[8];
#pragma unroll
  for (int r = 0; r < 8; r++) rinv[r] = 1.f / rsum[r];

  v8f cacc[4] = {vz, vz, vz, vz};
  const unsigned short* vb = vT + (size_t)h * 64 * N_;
  for (int j2 = 0; j2 < N_ / 16; j2 += 2) {
#pragma unroll
    for (int jj = 0; jj < 2; jj++) {
      int j = j2 + jj;
      v8f s = vz;
      s = wmma_bf(aq0, load_frag(kb, 3 * D_, j * 16 + rc, 0, hi), s);
      s = wmma_bf(aq1, load_frag(kb, 3 * D_, j * 16 + rc, 32, hi), s);
#pragma unroll
      for (int r = 0; r < 8; r++) {
        float p = __expf(s[r] - rmax[r]) * rinv[r];
        int mm = m0 + r + 8 * hi;
        atomicAdd(&attn_out[(size_t)mm * N_ + j * 16 + rc], p * 0.125f);
        pbuf[(r + 8 * hi) * 32 + jj * 16 + rc] = f2bf(p);
      }
    }
    __syncthreads();
    v16bf pa = load_frag(pbuf, 32, rc, 0, hi);
#pragma unroll
    for (int t = 0; t < 4; t++) {
      v16bf bv = load_frag(vb, N_, t * 16 + rc, j2 * 16, hi);
      cacc[t] = wmma_bf(pa, bv, cacc[t]);
    }
    __syncthreads();
  }
#pragma unroll
  for (int t = 0; t < 4; t++)
#pragma unroll
    for (int r = 0; r < 8; r++) {
      int mm = m0 + r + 8 * hi;
      ctx[(size_t)mm * D_ + h * 64 + t * 16 + rc] = f2bf(cacc[t][r]);
    }
}

// --------------------------------------------------- gate + residual combine
__global__ __launch_bounds__(128) void gate_kernel(
    const float* __restrict__ x, const float* __restrict__ xm, const float* __restrict__ xa,
    const float* __restrict__ gw, const float* __restrict__ gb, float* xc) {
  __shared__ float red[128];
  int row = blockIdx.x, t = threadIdx.x;
  size_t base = (size_t)row * D_;
  float s = 0.f;
#pragma unroll
  for (int i = 0; i < 4; i++) {
    int c = t * 4 + i;
    s += gw[c] * xm[base + c] + gw[D_ + c] * xa[base + c];
  }
  red[t] = s; __syncthreads();
  for (int o = 64; o > 0; o >>= 1) { if (t < o) red[t] += red[t + o]; __syncthreads(); }
  float g = 1.f / (1.f + __expf(-(red[0] + gb[0])));
#pragma unroll
  for (int i = 0; i < 4; i++) {
    int c = t * 4 + i;
    xc[base + c] = x[base + c] + g * xm[base + c] + (1.f - g) * xa[base + c];
  }
}

// ================================================================= host
extern "C" void kernel_launch(void* const* d_in, const int* in_sizes, int n_in,
                              void* d_out, int out_size, void* d_ws, size_t ws_size,
                              hipStream_t stream) {
  const float* x        = (const float*)d_in[0];
  const int*   ei       = (const int*)d_in[1];
  const float* ln1_g    = (const float*)d_in[2];
  const float* ln1_b    = (const float*)d_in[3];
  const float* ln2_g    = (const float*)d_in[4];
  const float* ln2_b    = (const float*)d_in[5];
  const float* ln3_g    = (const float*)d_in[6];
  const float* ln3_b    = (const float*)d_in[7];
  const float* in_proj_w  = (const float*)d_in[8];
  const float* in_proj_b  = (const float*)d_in[9];
  const float* out_proj_w = (const float*)d_in[10];
  const float* out_proj_b = (const float*)d_in[11];
  const float* msg_w    = (const float*)d_in[12];
  const float* msg_b    = (const float*)d_in[13];
  const float* gru_wih  = (const float*)d_in[14];
  const float* gru_whh  = (const float*)d_in[15];
  const float* gru_bih  = (const float*)d_in[16];
  const float* gru_bhh  = (const float*)d_in[17];
  const float* ffn_w1   = (const float*)d_in[18];
  const float* ffn_b1   = (const float*)d_in[19];
  const float* ffn_w2   = (const float*)d_in[20];
  const float* ffn_b2   = (const float*)d_in[21];
  const float* gate_w   = (const float*)d_in[22];
  const float* gate_b   = (const float*)d_in[23];

  const int E = in_sizes[1] / 2;
  const int* src = ei;
  const int* dst = ei + E;
  const int ND = N_ * D_;

  // ---- workspace carve
  char* wp = (char*)d_ws;
  auto alloc = [&](size_t bytes) -> void* {
    void* p = (void*)wp;
    wp += (bytes + 255) & ~(size_t)255;
    return p;
  };
  float*          xn1f   = (float*)alloc((size_t)ND * 4);
  unsigned short* xn1b   = (unsigned short*)alloc((size_t)ND * 2);
  unsigned short* xn2b   = (unsigned short*)alloc((size_t)ND * 2);
  unsigned short* xn3b   = (unsigned short*)alloc((size_t)ND * 2);
  float*          Al     = (float*)alloc((size_t)ND * 4);
  float*          Br     = (float*)alloc((size_t)ND * 4);
  float*          aggr   = (float*)alloc((size_t)ND * 4);
  unsigned short* aggrb  = (unsigned short*)alloc((size_t)ND * 2);
  float*          deg    = (float*)alloc((size_t)N_ * 4);
  float*          gi     = (float*)alloc((size_t)N_ * 3 * D_ * 4);
  float*          gh     = (float*)alloc((size_t)N_ * 3 * D_ * 4);
  float*          xmpnn  = (float*)alloc((size_t)ND * 4);
  unsigned short* qkvb   = (unsigned short*)alloc((size_t)N_ * 3 * D_ * 2);
  unsigned short* vT     = (unsigned short*)alloc((size_t)ND * 2);
  unsigned short* ctxb   = (unsigned short*)alloc((size_t)ND * 2);
  float*          xattn  = (float*)alloc((size_t)ND * 4);
  float*          xc     = (float*)alloc((size_t)ND * 4);
  unsigned short* hb     = (unsigned short*)alloc((size_t)N_ * 4 * D_ * 2);
  unsigned short* w_msg  = (unsigned short*)alloc((size_t)D_ * 2 * D_ * 2);
  unsigned short* w_inp  = (unsigned short*)alloc((size_t)3 * D_ * D_ * 2);
  unsigned short* w_outp = (unsigned short*)alloc((size_t)D_ * D_ * 2);
  unsigned short* w_wih  = (unsigned short*)alloc((size_t)3 * D_ * D_ * 2);
  unsigned short* w_whh  = (unsigned short*)alloc((size_t)3 * D_ * D_ * 2);
  unsigned short* w_f1   = (unsigned short*)alloc((size_t)4 * D_ * D_ * 2);
  unsigned short* w_f2   = (unsigned short*)alloc((size_t)D_ * 4 * D_ * 2);

  float* out_x    = (float*)d_out;              // [N, D]
  float* attn_out = (float*)d_out + (size_t)ND; // [N, N]

  const int T = 256;
  auto blks = [&](long long n) { return (unsigned)((n + T - 1) / T); };

  // ---- weight conversion f32 -> bf16
  cvt_kernel<<<blks(D_ * 2 * D_), T, 0, stream>>>(msg_w, w_msg, D_ * 2 * D_);
  cvt_kernel<<<blks(3 * D_ * D_), T, 0, stream>>>(in_proj_w, w_inp, 3 * D_ * D_);
  cvt_kernel<<<blks(D_ * D_), T, 0, stream>>>(out_proj_w, w_outp, D_ * D_);
  cvt_kernel<<<blks(3 * D_ * D_), T, 0, stream>>>(gru_wih, w_wih, 3 * D_ * D_);
  cvt_kernel<<<blks(3 * D_ * D_), T, 0, stream>>>(gru_whh, w_whh, 3 * D_ * D_);
  cvt_kernel<<<blks(4 * D_ * D_), T, 0, stream>>>(ffn_w1, w_f1, 4 * D_ * D_);
  cvt_kernel<<<blks(D_ * 4 * D_), T, 0, stream>>>(ffn_w2, w_f2, D_ * 4 * D_);

  // ---- zero accumulators (every call; deterministic)
  zero_kernel<<<blks(ND), T, 0, stream>>>(aggr, ND);
  zero_kernel<<<blks(N_), T, 0, stream>>>(deg, N_);
  zero_kernel<<<blks((long long)N_ * N_), T, 0, stream>>>(attn_out, (long long)N_ * N_);

  // ---- LN1 / LN2 fused (reads x once)
  ln_kernel<<<N_, 128, 0, stream>>>(x, ln1_g, ln1_b, xn1f, xn1b, ln2_g, ln2_b, xn2b);

  // ---- MPNN: Al = xn1@Wl^T ; Br = xn1@Wr^T + msg_b
  dim3 gD(D_ / 256, N_ / 32);          // O=512
  dim3 g3D(3 * D_ / 256, N_ / 32);     // O=1536
  dim3 g4D(4 * D_ / 256, N_ / 32);     // O=2048
  gemm_bf16_kernel<<<gD, 256, 0, stream>>>(xn1b, D_, w_msg, 2 * D_, nullptr, nullptr,
                                           Al, nullptr, D_, D_, 0);
  gemm_bf16_kernel<<<gD, 256, 0, stream>>>(xn1b, D_, w_msg + D_, 2 * D_, msg_b, nullptr,
                                           Br, nullptr, D_, D_, 0);
  deg_kernel<<<blks(E), T, 0, stream>>>(dst, deg, E);
  scatter_kernel<<<blks((long long)E * D_), T, 0, stream>>>(src, dst, Al, aggr, (long long)E * D_);
  aggr_fin_kernel<<<blks(ND), T, 0, stream>>>(aggr, Br, deg, aggrb, ND);

  // ---- GRU
  gemm_bf16_kernel<<<g3D, 256, 0, stream>>>(aggrb, D_, w_wih, D_, gru_bih, nullptr,
                                            gi, nullptr, 3 * D_, D_, 0);
  gemm_bf16_kernel<<<g3D, 256, 0, stream>>>(xn1b, D_, w_whh, D_, gru_bhh, nullptr,
                                            gh, nullptr, 3 * D_, D_, 0);
  gru_kernel<<<blks(ND), T, 0, stream>>>(gi, gh, xn1f, xmpnn, ND);

  // ---- attention: qkv (q pre-scaled 1/8), V transpose, streaming softmax + ctx
  gemm_bf16_kernel<<<g3D, 256, 0, stream>>>(xn2b, D_, w_inp, D_, in_proj_b, nullptr,
                                            nullptr, qkvb, 3 * D_, D_, 2);
  vt_kernel<<<blks(ND), T, 0, stream>>>(qkvb, vT, ND);
  attn_kernel<<<dim3(N_ / 16, H_), 32, 0, stream>>>(qkvb, vT, attn_out, ctxb);
  gemm_bf16_kernel<<<gD, 256, 0, stream>>>(ctxb, D_, w_outp, D_, out_proj_b, nullptr,
                                           xattn, nullptr, D_, D_, 0);

  // ---- gated combine + residual
  gate_kernel<<<N_, 128, 0, stream>>>(x, xmpnn, xattn, gate_w, gate_b, xc);

  // ---- FFN with fused GELU and residual into d_out
  ln_kernel<<<N_, 128, 0, stream>>>(xc, ln3_g, ln3_b, nullptr, xn3b,
                                    nullptr, nullptr, nullptr);
  gemm_bf16_kernel<<<g4D, 256, 0, stream>>>(xn3b, D_, w_f1, D_, ffn_b1, nullptr,
                                            nullptr, hb, 4 * D_, D_, 1);
  gemm_bf16_kernel<<<gD, 256, 0, stream>>>(hb, 4 * D_, w_f2, 4 * D_, ffn_b2, xc,
                                           out_x, nullptr, D_, 4 * D_, 0);
}
